// KeypointBSplineNet_48687749267937
// MI455X (gfx1250) — compile-verified, matrix operand order checked by reading, not run on hardware
//
#include <hip/hip_runtime.h>
#include <hip/hip_bf16.h>
#include <math.h>

// ---------------------------------------------------------------------------
// KeypointBSplineNet fused inference for gfx1250 (MI455X)
//   - all GEMMs via v_wmma_f32_16x16x32_bf16 (f32 accumulate)
//   - weights pre-packed to bf16 WMMA B-fragment layout in d_ws
//   - activations ping-pong through LDS (bank-conflict-free padded strides)
//   - spline + Cobb geometry done per-lane after the matrix phase
// ---------------------------------------------------------------------------

typedef __bf16 bf16;
typedef __attribute__((ext_vector_type(16))) __bf16 v16bf;
typedef __attribute__((ext_vector_type(8)))  float  v8f;

#define NB 32768              // batch
#define SA 520                // LDS stride (bf16 elems) bufA: 1040B -> 4-bank shift/row
#define SB 264                // LDS stride bufB: 528B  -> 4-bank shift/row

// Packed-weight workspace offsets (bf16 elements); each tile = 512 bf16.
// tiles laid out as [ntile][ktile], tile idx = nt*Ktiles + kt
enum {
  OFF_W1  = 0,        // K=68->96 (3 ktiles), N=256 (16 ntiles)  : 48*512
  OFF_W2  = 24576,    // K=256 (8),  N=512 (32)                  : 256*512
  OFF_W3  = 155648,   // K=512 (16), N=256 (16)                  : 256*512
  OFF_WC1 = 286720,   // K=256 (8),  N=128 (8)                   : 64*512
  OFF_WC2 = 319488,   // K=128 (4),  N=20->32 (2)                : 8*512
  OFF_WK1 = 323584,   // K=256 (8),  N=128 (8)                   : 64*512
  OFF_WK2 = 356352,   // K=128 (4),  N=6->16 (1)                 : 4*512
  WS_BF16_TOTAL = 358400
};

union AFrag { v16bf v; uint4 q[2]; };

// ---------------------------------------------------------------------------
// Weight packing: fp32 [K,N] row-major -> bf16 WMMA B-fragments.
// One 32-lane block per 16x32 tile. Lane L holds column N = nt*16 + (L&15);
// element j maps to kk = j + (j>=8 ? 8 : 0) + (L>=16 ? 8 : 0) within the ktile
// (matches the ISA 16-bit operand layout). Out-of-range K/N -> 0.
// ---------------------------------------------------------------------------
__global__ void pack_weights(const float* __restrict__ w, bf16* __restrict__ out,
                             int K, int N, int Ktiles) {
  int bid  = blockIdx.x;
  int kt   = bid % Ktiles;
  int nt   = bid / Ktiles;
  int lane = threadIdx.x;
  int hf   = lane >> 4;
  int n    = nt * 16 + (lane & 15);
  bf16* dst = out + (size_t)bid * 512 + lane * 16;
#pragma unroll
  for (int j = 0; j < 16; ++j) {
    int kk = j + ((j >= 8) ? 8 : 0) + hf * 8;
    int k  = kt * 32 + kk;
    float v = (k < K && n < N) ? w[(size_t)k * N + n] : 0.0f;
    dst[j] = (bf16)v;
  }
}

// A-fragment from LDS (row-major bf16, padded stride): two ds_load_b128.
__device__ __forceinline__ v16bf load_a_lds(const bf16* buf, int stride,
                                            int colOff, int kt, int lane) {
  int row = lane & 15;
  int hf  = lane >> 4;
  const bf16* p = buf + row * stride + colOff + kt * 32 + hf * 8;
  AFrag f;
  f.q[0] = *(const uint4*)(p);        // j0..7  -> kk +0..7
  f.q[1] = *(const uint4*)(p + 16);   // j8..15 -> kk +16..23
  return f.v;
}

// B-fragment from packed workspace: 32 contiguous bytes per lane.
__device__ __forceinline__ v16bf load_b_glb(const bf16* w, int lane) {
  const bf16* p = w + lane * 16;
  AFrag f;
  f.q[0] = *(const uint4*)(p);
  f.q[1] = *(const uint4*)(p + 8);
  return f.v;
}

// relu(in @ W + b) -> LDS bf16 (all Ntiles of 16 cols)
__device__ __forceinline__ void gemm_relu(const bf16* inB, int inStride,
                                          bf16* outB, int outStride, int outColOff,
                                          const bf16* wt, const float* bias,
                                          int Ntiles, int Ktiles, int lane) {
  int nl = lane & 15;
  int hf = lane >> 4;
  for (int nt = 0; nt < Ntiles; ++nt) {
    v8f c = {};
    for (int kt = 0; kt < Ktiles; ++kt) {
      v16bf a = load_a_lds(inB, inStride, 0, kt, lane);
      v16bf b = load_b_glb(wt + (size_t)(nt * Ktiles + kt) * 512, lane);
      c = __builtin_amdgcn_wmma_f32_16x16x32_bf16(false, a, false, b,
                                                  (short)0, c, false, false);
    }
    int n = nt * 16 + nl;
    float bv = bias[n];
#pragma unroll
    for (int r = 0; r < 8; ++r) {
      int m = r + hf * 8;
      float v = c[r] + bv;
      v = v > 0.f ? v : 0.f;
      outB[m * outStride + outColOff + n] = (bf16)v;
    }
  }
}

// head: in @ W + b (optional clamp[0,1]) -> f32 LDS stage + global out
__device__ __forceinline__ void head_out(const bf16* inB, int inStride, int inColOff,
                                         const bf16* wt, const float* bias,
                                         int Nvalid, int Ntiles, int Ktiles,
                                         bool clip01, float* stage,
                                         float* outG, int rows0, int lane) {
  int nl = lane & 15;
  int hf = lane >> 4;
  for (int nt = 0; nt < Ntiles; ++nt) {
    v8f c = {};
    for (int kt = 0; kt < Ktiles; ++kt) {
      v16bf a = load_a_lds(inB, inStride, inColOff, kt, lane);
      v16bf b = load_b_glb(wt + (size_t)(nt * Ktiles + kt) * 512, lane);
      c = __builtin_amdgcn_wmma_f32_16x16x32_bf16(false, a, false, b,
                                                  (short)0, c, false, false);
    }
    int n = nt * 16 + nl;
    float bv = (n < Nvalid) ? bias[n] : 0.f;
#pragma unroll
    for (int r = 0; r < 8; ++r) {
      int m = r + hf * 8;
      float v = c[r] + bv;
      if (clip01) v = v < 0.f ? 0.f : (v > 1.f ? 1.f : v);
      if (n < Nvalid) {
        stage[m * Nvalid + n] = v;
        outG[(size_t)(rows0 + m) * Nvalid + n] = v;
      }
    }
  }
}

__global__ __launch_bounds__(64) void fused_net(
    const float* __restrict__ kp, const float* __restrict__ knots,
    const float* __restrict__ cpin,
    const float* __restrict__ b1, const float* __restrict__ b2,
    const float* __restrict__ b3, const float* __restrict__ bc1,
    const float* __restrict__ bc2, const float* __restrict__ bk1,
    const float* __restrict__ bk2,
    const bf16* __restrict__ ws, float* __restrict__ out) {
  __shared__ __align__(16) bf16  sA[2][16 * SA];
  __shared__ __align__(16) bf16  sB[2][16 * SB];
  __shared__ float sCP[2][16 * 20];
  __shared__ float sKN[2][16 * 6];

  int tid  = threadIdx.x;
  int wv   = tid >> 5;
  int lane = tid & 31;
  int rows0 = blockIdx.x * 32 + wv * 16;
  bf16* A  = sA[wv];
  bf16* Bb = sB[wv];

  // load x = keypoints[rows0..rows0+15, 0:68] -> bufA, zero-pad K to 96
  for (int t = lane; t < 16 * 96; t += 32) {
    int r = t / 96, c = t % 96;
    float v = (c < 68) ? kp[(size_t)(rows0 + r) * 68 + c] : 0.f;
    A[r * SA + c] = (bf16)v;
  }
  __syncthreads();

  gemm_relu(A, SA, Bb, SB, 0,   ws + OFF_W1,  b1, 16, 3,  lane);  // 68->256
  __syncthreads();
  gemm_relu(Bb, SB, A, SA, 0,   ws + OFF_W2,  b2, 32, 8,  lane);  // 256->512
  __syncthreads();
  gemm_relu(A, SA, Bb, SB, 0,   ws + OFF_W3,  b3, 16, 16, lane);  // 512->256 (feat)
  __syncthreads();
  gemm_relu(Bb, SB, A, SA, 0,   ws + OFF_WC1, bc1, 8, 8,  lane);  // feat->hc1 (cols 0..127)
  gemm_relu(Bb, SB, A, SA, 128, ws + OFF_WK1, bk1, 8, 8,  lane);  // feat->hk1 (cols 128..255)
  __syncthreads();
  // cp head: hc1 @ wc2 + bc2 -> deta_cp [B,20] at out + 3*NB
  head_out(A, SA, 0,   ws + OFF_WC2, bc2, 20, 2, 4, false, sCP[wv],
           out + (size_t)3 * NB, rows0, lane);
  // knot head: clamp01(hk1 @ wk2 + bk2) -> deta_knots [B,6] at out + 23*NB
  head_out(A, SA, 128, ws + OFF_WK2, bk2, 6, 1, 4, true, sKN[wv],
           out + (size_t)23 * NB, rows0, lane);
  __syncthreads();

  // ---- B-spline eval + Cobb angles: lane L<16 handles row rows0+L ----
  if (lane < 16) {
    int row = rows0 + lane;
    float u[14];
#pragma unroll
    for (int i = 0; i < 14; ++i) {
      float d = (i >= 4 && i < 10) ? sKN[wv][lane * 6 + (i - 4)] : 0.f;
      u[i] = knots[(size_t)row * 14 + i] + d;
    }
    float cpx[10], cpy[10];
#pragma unroll
    for (int i = 0; i < 10; ++i) {
      cpx[i] = cpin[(size_t)row * 20 + 2 * i]     + sCP[wv][lane * 20 + 2 * i];
      cpy[i] = cpin[(size_t)row * 20 + 2 * i + 1] + sCP[wv][lane * 20 + 2 * i + 1];
    }
    float ang[33];
    float prevx = 0.f, prevy = 0.f;
    for (int q = 0; q < 34; ++q) {
      float uq = (float)q * (1.0f / 33.0f);
      float ulim = u[13] - 1e-6f;
      uq = uq < ulim ? uq : ulim;
      float Nb[13];
#pragma unroll
      for (int i = 0; i < 13; ++i)
        Nb[i] = (uq >= u[i] && uq < u[i + 1]) ? 1.f : 0.f;
#pragma unroll
      for (int p = 1; p <= 3; ++p) {
        for (int i = 0; i < 13 - p; ++i) {
          float ld = u[i + p] - u[i];
          float lt = ld > 0.f ? (uq - u[i]) / ld : 0.f;
          float rd = u[i + p + 1] - u[i + 1];
          float rt = rd > 0.f ? (u[i + p + 1] - uq) / rd : 0.f;
          Nb[i] = lt * Nb[i] + rt * Nb[i + 1];
        }
      }
      float px = 0.f, py = 0.f;
#pragma unroll
      for (int i = 0; i < 10; ++i) { px += Nb[i] * cpx[i]; py += Nb[i] * cpy[i]; }
      if (q > 0)
        ang[q - 1] = atan2f(px - prevx, py - prevy) * 57.29577951308232f;
      prevx = px; prevy = py;
    }
    int imax = 0, imin = 0;
    for (int i = 1; i < 33; ++i) {
      if (ang[i] > ang[imax]) imax = i;   // first max (strict >)
      if (ang[i] < ang[imin]) imin = i;   // first min
    }
    float mt = ang[imax] - ang[imin];
    int lo = imax < imin ? imax : imin;
    int hi = imax > imin ? imax : imin;
    float mx = -3.4e38f, mn = 3.4e38f;
    for (int i = 0; i <= lo; ++i) { mx = ang[i] > mx ? ang[i] : mx; mn = ang[i] < mn ? ang[i] : mn; }
    float pt = mx - mn;
    mx = -3.4e38f; mn = 3.4e38f;
    for (int i = hi; i < 33; ++i) { mx = ang[i] > mx ? ang[i] : mx; mn = ang[i] < mn ? ang[i] : mn; }
    float tl = mx - mn;
    out[(size_t)row * 3 + 0] = pt;
    out[(size_t)row * 3 + 1] = mt;
    out[(size_t)row * 3 + 2] = tl;
  }
}

extern "C" void kernel_launch(void* const* d_in, const int* in_sizes, int n_in,
                              void* d_out, int out_size, void* d_ws, size_t ws_size,
                              hipStream_t stream) {
  const float* kp    = (const float*)d_in[0];
  const float* knots = (const float*)d_in[1];
  const float* cp    = (const float*)d_in[2];
  const float* w1  = (const float*)d_in[3];  const float* b1  = (const float*)d_in[4];
  const float* w2  = (const float*)d_in[5];  const float* b2  = (const float*)d_in[6];
  const float* w3  = (const float*)d_in[7];  const float* b3  = (const float*)d_in[8];
  const float* wc1 = (const float*)d_in[9];  const float* bc1 = (const float*)d_in[10];
  const float* wc2 = (const float*)d_in[11]; const float* bc2 = (const float*)d_in[12];
  const float* wk1 = (const float*)d_in[13]; const float* bk1 = (const float*)d_in[14];
  const float* wk2 = (const float*)d_in[15]; const float* bk2 = (const float*)d_in[16];
  bf16*  ws  = (bf16*)d_ws;
  float* out = (float*)d_out;

  // pack fp32 weights -> bf16 WMMA fragments (deterministic every call)
  pack_weights<<<48,  32, 0, stream>>>(w1,  ws + OFF_W1,  68,  256, 3);
  pack_weights<<<256, 32, 0, stream>>>(w2,  ws + OFF_W2,  256, 512, 8);
  pack_weights<<<256, 32, 0, stream>>>(w3,  ws + OFF_W3,  512, 256, 16);
  pack_weights<<<64,  32, 0, stream>>>(wc1, ws + OFF_WC1, 256, 128, 8);
  pack_weights<<<8,   32, 0, stream>>>(wc2, ws + OFF_WC2, 128, 20,  4);
  pack_weights<<<64,  32, 0, stream>>>(wk1, ws + OFF_WK1, 256, 128, 8);
  pack_weights<<<4,   32, 0, stream>>>(wk2, ws + OFF_WK2, 128, 6,   4);

  // fused MLP + spline + Cobb: 1024 blocks x 64 threads (2 waves, 16 rows/wave)
  fused_net<<<NB / 32, 64, 0, stream>>>(kp, knots, cp, b1, b2, b3, bc1, bc2,
                                        bk1, bk2, ws, out);
}
